// Mamba2Block_59261958750642
// MI455X (gfx1250) — compile-verified, matrix operand order checked by reading, not run on hardware
//
#include <hip/hip_runtime.h>
#include <math.h>

#define DMODEL  2048
#define DINNER  4096
#define NHEADS  64
#define HEADDIM 64
#define DSTATE  64
#define DCONV   4
#define DPROJ   8384          // 2*DINNER + 2*DSTATE + NHEADS
#define BATCH   2
#define SEQLEN  4096
#define NTOK    (BATCH*SEQLEN)   // 8192
#define CHUNK   64
#define NCHUNK  (SEQLEN/CHUNK)   // 64
#define RMSEPS  1e-6f

typedef __attribute__((ext_vector_type(16))) __bf16 v16bf;
typedef __attribute__((ext_vector_type(8)))  __bf16 bf16x8;
typedef __attribute__((ext_vector_type(8)))  float  v8f;

// ---------------------------------------------------------------------------
// WMMA fragment loader: row-major bf16 source, 16x32 tile (A layout; B uses the
// same pattern against a pre-transposed [N][K] matrix).
// lanes 0-15: row = row0+lane,    K = k0+{0..7} and k0+{16..23}
// lanes16-31: row = row0+lane-16, K = k0+{8..15} and k0+{24..31}
// ---------------------------------------------------------------------------
__device__ __forceinline__ v16bf load_frag(const __bf16* __restrict__ base,
                                           int stride, int row0, int k0) {
  int lane = threadIdx.x & 31;
  int r  = row0 + (lane & 15);
  int kb = k0 + ((lane >> 4) << 3);
  const __bf16* p = base + (size_t)r * stride + kb;
  bf16x8 lo = *(const bf16x8*)(p);
  bf16x8 hi = *(const bf16x8*)(p + 16);
  v16bf f;
#pragma unroll
  for (int i = 0; i < 8; ++i) { f[i] = lo[i]; f[8 + i] = hi[i]; }
  return f;
}

#define WMMA_BF16(A, B, C) \
  __builtin_amdgcn_wmma_f32_16x16x32_bf16(false, (A), false, (B), (short)0, (C), false, false)

// issue a wave-wide async 16B/lane global->LDS copy (ASYNCcnt tracked)
__device__ __forceinline__ void async_copy_b128(unsigned lds_byte_addr,
                                                const void* gaddr) {
  unsigned long long ga = (unsigned long long)gaddr;
  asm volatile("global_load_async_to_lds_b128 %0, %1, off"
               :: "v"(lds_byte_addr), "v"(ga) : "memory");
}
__device__ __forceinline__ void async_wait0() {
  asm volatile("s_wait_asynccnt 0" ::: "memory");
}

// ---------------------------------------------------------------------------
// Generic bf16 GEMM: C[M,N] = A[M,K] * Bt[N,K]^T, f32 accumulate.
// 128 threads = 4 waves; block tile 64x64; wave tile 32x32 (2x2 WMMA frags).
// Operands are L2-resident (both fit in 192MB); prefetch hides first-touch.
// ---------------------------------------------------------------------------
__global__ __launch_bounds__(128) void gemm_bf16_wmma(
    const __bf16* __restrict__ A, const __bf16* __restrict__ Bt,
    float* __restrict__ C, int M, int N, int K) {
  int lane = threadIdx.x & 31;
  int wave = threadIdx.x >> 5;
  int wm = wave & 1, wn = wave >> 1;
  int m0 = blockIdx.y * 64 + wm * 32;
  int n0 = blockIdx.x * 64 + wn * 32;

  const __bf16* pfa = A  + (size_t)(m0 + (lane & 31)) * K;
  const __bf16* pfb = Bt + (size_t)(n0 + (lane & 31)) * K;

  v8f acc[2][2] = {};
  for (int kk = 0; kk < K; kk += 32) {
    if (kk + 160 <= K) {   // prefetch 4 k-steps ahead on both row streams
      __builtin_prefetch(pfa + kk + 128, 0, 0);
      __builtin_prefetch(pfb + kk + 128, 0, 0);
    }
    v16bf a0 = load_frag(A, K, m0,      kk);
    v16bf a1 = load_frag(A, K, m0 + 16, kk);
    v16bf b0 = load_frag(Bt, K, n0,      kk);
    v16bf b1 = load_frag(Bt, K, n0 + 16, kk);
    acc[0][0] = WMMA_BF16(a0, b0, acc[0][0]);
    acc[0][1] = WMMA_BF16(a0, b1, acc[0][1]);
    acc[1][0] = WMMA_BF16(a1, b0, acc[1][0]);
    acc[1][1] = WMMA_BF16(a1, b1, acc[1][1]);
  }
  // C/D layout: VGPR j -> M=j (lanes 0-15) / M=j+8 (lanes 16-31); N = lane&15
  int rbase = (lane >> 4) << 3;
  int col   = lane & 15;
#pragma unroll
  for (int mi = 0; mi < 2; ++mi)
#pragma unroll
    for (int ni = 0; ni < 2; ++ni)
#pragma unroll
      for (int j = 0; j < 8; ++j) {
        int m = m0 + mi * 16 + rbase + j;
        int n = n0 + ni * 16 + col;
        C[(size_t)m * N + n] = acc[mi][ni][j];
      }
}

// ---------------------------------------------------------------------------
// f32 -> bf16 copy
// ---------------------------------------------------------------------------
__global__ void cvt_bf16_kernel(const float* __restrict__ src,
                                __bf16* __restrict__ dst, size_t n) {
  size_t i = (size_t)blockIdx.x * blockDim.x + threadIdx.x;
  if (i < n) dst[i] = (__bf16)src[i];
}

// f32 [R][Ccols] -> bf16 transposed [Ccols][R]
__global__ void transpose_bf16_kernel(const float* __restrict__ W,
                                      __bf16* __restrict__ Wt, int R, int Ccols) {
  size_t i = (size_t)blockIdx.x * blockDim.x + threadIdx.x;
  if (i >= (size_t)R * Ccols) return;
  int r = (int)(i / Ccols);
  int c = (int)(i % Ccols);
  Wt[(size_t)c * R + r] = (__bf16)W[i];
}

// ---------------------------------------------------------------------------
// depthwise causal conv1d (K=4) + SiLU on the x slice of zxBCdt
// ---------------------------------------------------------------------------
__global__ void conv_silu_kernel(const float* __restrict__ zx,
                                 const float* __restrict__ cw,   // [4][DINNER]
                                 const float* __restrict__ cb,   // [DINNER]
                                 float* __restrict__ vf, __bf16* __restrict__ vb) {
  size_t idx = (size_t)blockIdx.x * blockDim.x + threadIdx.x;
  if (idx >= (size_t)NTOK * DINNER) return;
  int d = (int)(idx % DINNER);
  size_t row = idx / DINNER;           // b*L + l
  int l = (int)(row % SEQLEN);
  float s = cb[d];
#pragma unroll
  for (int j = 0; j < DCONV; ++j) {
    int ll = l - (DCONV - 1) + j;
    if (ll >= 0)
      s += zx[(row + (size_t)(ll - l)) * DPROJ + DINNER + d] * cw[j * DINNER + d];
  }
  float y = s / (1.f + __expf(-s));    // silu
  vf[idx] = y;
  vb[idx] = (__bf16)y;
}

// ---------------------------------------------------------------------------
// per-token: elu+1 on B/C projections -> k/q bf16; softplus dt -> log-decay la
// grid = NTOK, block = 64 (DSTATE == NHEADS == 64)
// ---------------------------------------------------------------------------
__global__ void qkla_kernel(const float* __restrict__ zx,
                            const float* __restrict__ A_log,
                            const float* __restrict__ dt_bias,
                            __bf16* __restrict__ qb, __bf16* __restrict__ kb,
                            float* __restrict__ la) {
  size_t row = blockIdx.x;
  int t = threadIdx.x;                 // 0..63
  const float* rz = zx + row * DPROJ;
  float bp = rz[2 * DINNER + t];
  float cp = rz[2 * DINNER + DSTATE + t];
  kb[row * DSTATE + t] = (__bf16)(bp > 0.f ? bp + 1.f : __expf(bp));
  qb[row * DSTATE + t] = (__bf16)(cp > 0.f ? cp + 1.f : __expf(cp));
  float dl = rz[2 * DINNER + 2 * DSTATE + t] + dt_bias[t];
  float dt = dl > 20.f ? dl : log1pf(__expf(dl));
  la[row * NHEADS + t] = -dt * __expf(A_log[t]);
}

// ---------------------------------------------------------------------------
// SSD chunked scan: one block per (batch, head); 128 threads = 4 waves.
// q/k chunk tiles are async-copied to LDS (each is consumed by 2 matmuls).
// State h (64x64) lives in LDS (f32 + transposed bf16 copy for WMMA B-frags).
// ---------------------------------------------------------------------------
__global__ __launch_bounds__(128) void ssd_kernel(
    const __bf16* __restrict__ qb, const __bf16* __restrict__ kb,
    const __bf16* __restrict__ vb, const float* __restrict__ vf,
    const float* __restrict__ la, const float* __restrict__ Dp,
    float* __restrict__ y) {
  int bh = blockIdx.x;                  // 0..BATCH*NHEADS-1
  int b  = bh >> 6;
  int hd = bh & 63;
  int tid  = threadIdx.x;
  int lane = tid & 31, wave = tid >> 5;
  int wm = wave & 1, wn = wave >> 1;
  int m0  = wm * 32;
  int n0w = wn * 32;
  int rbase = (lane >> 4) << 3;
  int col   = lane & 15;

  __shared__ __align__(16) float  cs[CHUNK];
  __shared__ __align__(16) float  wsc[CHUNK];    // exp(total - cs_s)
  __shared__ __align__(16) float  ecs[CHUNK];    // exp(cs_s)
  __shared__ float etot_s;
  __shared__ __align__(16) __bf16 qs[CHUNK][DSTATE];      // async-staged q tile
  __shared__ __align__(16) __bf16 ks[CHUNK][DSTATE];      // async-staged k tile
  __shared__ __align__(16) __bf16 attb[CHUNK][CHUNK];
  __shared__ __align__(16) __bf16 vT[HEADDIM][CHUNK];     // vT[p][s]
  __shared__ __align__(16) __bf16 kTs[DSTATE][CHUNK];     // scaled k^T [n][s]
  __shared__ __align__(16) __bf16 hbT[HEADDIM][DSTATE];   // h^T bf16 [p][n]
  __shared__ __align__(16) float  hf[DSTATE][HEADDIM];    // h f32 [n][p]

  for (int i = tid; i < DSTATE * HEADDIM; i += 128) {
    hf[i >> 6][i & 63]  = 0.f;
    hbT[i >> 6][i & 63] = (__bf16)0.f;
  }
  float Dh = Dp[hd];
  unsigned lq = (unsigned)(unsigned long long)&qs[0][0];
  unsigned lk = (unsigned)(unsigned long long)&ks[0][0];

  for (int c = 0; c < NCHUNK; ++c) {
    size_t rowbase = (size_t)b * SEQLEN + (size_t)c * CHUNK;
    const __bf16* qblk = qb + rowbase * DSTATE;   // contiguous 64x64 bf16 tile
    const __bf16* kblk = kb + rowbase * DSTATE;

    // ---- kick off async global->LDS copies of q/k tiles (8KB each) ----
#pragma unroll
    for (int it = 0; it < 4; ++it) {
      int e = tid + it * 128;                     // 16B granules, 512 total
      async_copy_b128(lq + e * 16, (const char*)qblk + e * 16);
      async_copy_b128(lk + e * 16, (const char*)kblk + e * 16);
    }

    // ---- decay prefix sums (overlapped with async copies) ----
    if (tid < CHUNK) cs[tid] = la[(rowbase + tid) * NHEADS + hd];
    __syncthreads();
    if (tid == 0) {
      float s = 0.f;
      for (int i = 0; i < CHUNK; ++i) { s += cs[i]; cs[i] = s; }
      etot_s = __expf(s);
    }
    __syncthreads();
    if (tid < CHUNK) {
      float tot = cs[CHUNK - 1];
      wsc[tid] = __expf(tot - cs[tid]);
      ecs[tid] = __expf(cs[tid]);
    }
    // stage v^T (no dependence on wsc)
    for (int i = tid; i < CHUNK * HEADDIM; i += 128) {
      int s = i >> 6, p = i & 63;
      vT[p][s] = vb[(rowbase + s) * DINNER + hd * HEADDIM + p];
    }
    async_wait0();            // q/k tiles landed in LDS (this wave's copies)
    __syncthreads();          // all waves' copies + wsc/vT visible
    // stage scaled k^T (reads k tile from LDS now)
    for (int i = tid; i < CHUNK * DSTATE; i += 128) {
      int n = i >> 6, s = i & 63;
      kTs[n][s] = (__bf16)((float)ks[s][n] * wsc[s]);
    }
    __syncthreads();

    // ---- att = q @ k^T ; y_inter = q @ h_prev  (all operands in LDS) ----
    v8f aAtt[2][2] = {};
    v8f aInt[2][2] = {};
#pragma unroll
    for (int kk = 0; kk < DSTATE; kk += 32) {
      v16bf a0 = load_frag(&qs[0][0], DSTATE, m0,      kk);
      v16bf a1 = load_frag(&qs[0][0], DSTATE, m0 + 16, kk);
      v16bf b0 = load_frag(&ks[0][0], DSTATE, n0w,      kk);
      v16bf b1 = load_frag(&ks[0][0], DSTATE, n0w + 16, kk);
      aAtt[0][0] = WMMA_BF16(a0, b0, aAtt[0][0]);
      aAtt[0][1] = WMMA_BF16(a0, b1, aAtt[0][1]);
      aAtt[1][0] = WMMA_BF16(a1, b0, aAtt[1][0]);
      aAtt[1][1] = WMMA_BF16(a1, b1, aAtt[1][1]);
      v16bf h0 = load_frag(&hbT[0][0], DSTATE, n0w,      kk);
      v16bf h1 = load_frag(&hbT[0][0], DSTATE, n0w + 16, kk);
      aInt[0][0] = WMMA_BF16(a0, h0, aInt[0][0]);
      aInt[0][1] = WMMA_BF16(a0, h1, aInt[0][1]);
      aInt[1][0] = WMMA_BF16(a1, h0, aInt[1][0]);
      aInt[1][1] = WMMA_BF16(a1, h1, aInt[1][1]);
    }
    // apply causal decay mask, write att as bf16 to LDS
#pragma unroll
    for (int mi = 0; mi < 2; ++mi)
#pragma unroll
      for (int ni = 0; ni < 2; ++ni)
#pragma unroll
        for (int j = 0; j < 8; ++j) {
          int s = m0 + mi * 16 + rbase + j;
          int t = n0w + ni * 16 + col;
          float v = (t <= s) ? aAtt[mi][ni][j] * __expf(cs[s] - cs[t]) : 0.f;
          attb[s][t] = (__bf16)v;
        }
    __syncthreads();   // attb ready; all lanes done reading hbT

    // ---- y_intra = att @ v ; states = kTs @ v  (B-frags from vT[p][s]) ----
    v8f aY[2][2]  = {};
    v8f aSt[2][2] = {};
#pragma unroll
    for (int kk = 0; kk < CHUNK; kk += 32) {
      v16bf t0 = load_frag(&attb[0][0], CHUNK, m0,      kk);
      v16bf t1 = load_frag(&attb[0][0], CHUNK, m0 + 16, kk);
      v16bf k0 = load_frag(&kTs[0][0],  CHUNK, m0,      kk);
      v16bf k1 = load_frag(&kTs[0][0],  CHUNK, m0 + 16, kk);
      v16bf v0 = load_frag(&vT[0][0],   CHUNK, n0w,      kk);
      v16bf v1 = load_frag(&vT[0][0],   CHUNK, n0w + 16, kk);
      aY[0][0] = WMMA_BF16(t0, v0, aY[0][0]);
      aY[0][1] = WMMA_BF16(t0, v1, aY[0][1]);
      aY[1][0] = WMMA_BF16(t1, v0, aY[1][0]);
      aY[1][1] = WMMA_BF16(t1, v1, aY[1][1]);
      aSt[0][0] = WMMA_BF16(k0, v0, aSt[0][0]);
      aSt[0][1] = WMMA_BF16(k0, v1, aSt[0][1]);
      aSt[1][0] = WMMA_BF16(k1, v0, aSt[1][0]);
      aSt[1][1] = WMMA_BF16(k1, v1, aSt[1][1]);
    }

    // ---- emit y = y_intra + exp(cs_s)*y_inter + D*v ----
    float etot = etot_s;
#pragma unroll
    for (int mi = 0; mi < 2; ++mi)
#pragma unroll
      for (int ni = 0; ni < 2; ++ni)
#pragma unroll
        for (int j = 0; j < 8; ++j) {
          int s = m0 + mi * 16 + rbase + j;
          int p = n0w + ni * 16 + col;
          size_t oidx = (rowbase + s) * DINNER + hd * HEADDIM + p;
          y[oidx] = aY[mi][ni][j] + ecs[s] * aInt[mi][ni][j] + Dh * vf[oidx];
        }
    __syncthreads();   // done with vT/attb/kTs/qs/ks before restage

    // ---- h = exp(total)*h + states ----
#pragma unroll
    for (int mi = 0; mi < 2; ++mi)
#pragma unroll
      for (int ni = 0; ni < 2; ++ni)
#pragma unroll
        for (int j = 0; j < 8; ++j) {
          int n = m0 + mi * 16 + rbase + j;
          int p = n0w + ni * 16 + col;
          float hn = etot * hf[n][p] + aSt[mi][ni][j];
          hf[n][p]  = hn;
          hbT[p][n] = (__bf16)hn;
        }
    __syncthreads();
  }
}

// ---------------------------------------------------------------------------
// RMSNorm over DINNER + silu(z) gate -> bf16 for the out_proj GEMM
// grid = NTOK, block = 256
// ---------------------------------------------------------------------------
__global__ __launch_bounds__(256) void rmsnorm_gate_kernel(
    const float* __restrict__ y, const float* __restrict__ zx,
    const float* __restrict__ scale, __bf16* __restrict__ yb) {
  int row = blockIdx.x;
  __shared__ float red[256];
  const float* yr = y + (size_t)row * DINNER;
  float ss = 0.f;
  for (int d = threadIdx.x; d < DINNER; d += 256) { float v = yr[d]; ss += v * v; }
  red[threadIdx.x] = ss;
  __syncthreads();
  for (int off = 128; off > 0; off >>= 1) {
    if (threadIdx.x < off) red[threadIdx.x] += red[threadIdx.x + off];
    __syncthreads();
  }
  float inv = rsqrtf(red[0] / (float)DINNER + RMSEPS);
  const float* zr = zx + (size_t)row * DPROJ;
  for (int d = threadIdx.x; d < DINNER; d += 256) {
    float z = zr[d];
    float g = z / (1.f + __expf(-z));
    yb[(size_t)row * DINNER + d] = (__bf16)(yr[d] * inv * scale[d] * g);
  }
}

// ---------------------------------------------------------------------------
extern "C" void kernel_launch(void* const* d_in, const int* in_sizes, int n_in,
                              void* d_out, int out_size, void* d_ws, size_t ws_size,
                              hipStream_t stream) {
  const float* u        = (const float*)d_in[0];
  const float* in_w     = (const float*)d_in[1];
  const float* conv_w   = (const float*)d_in[2];
  const float* conv_b   = (const float*)d_in[3];
  const float* A_log    = (const float*)d_in[4];
  const float* dt_bias  = (const float*)d_in[5];
  const float* D_param  = (const float*)d_in[6];
  const float* nscale   = (const float*)d_in[7];
  const float* out_w    = (const float*)d_in[8];
  float* out            = (float*)d_out;

  char* wp = (char*)d_ws;
  auto take = [&](size_t bytes) {
    char* r = wp;
    wp += (bytes + 255) & ~(size_t)255;
    return (void*)r;
  };
  __bf16* ub   = (__bf16*)take((size_t)NTOK * DMODEL * 2);
  __bf16* wtb  = (__bf16*)take((size_t)DPROJ * DMODEL * 2);    // in_proj^T bf16
  __bf16* wtb2 = (__bf16*)take((size_t)DMODEL * DINNER * 2);   // out_proj^T bf16
  float*  zx   = (float*) take((size_t)NTOK * DPROJ * 4);
  float*  vf   = (float*) take((size_t)NTOK * DINNER * 4);
  __bf16* vb   = (__bf16*)take((size_t)NTOK * DINNER * 2);
  __bf16* qb   = (__bf16*)take((size_t)NTOK * DSTATE * 2);
  __bf16* kb   = (__bf16*)take((size_t)NTOK * DSTATE * 2);
  float*  la   = (float*) take((size_t)NTOK * NHEADS * 4);
  float*  yf   = (float*) take((size_t)NTOK * DINNER * 4);
  __bf16* yb   = (__bf16*)take((size_t)NTOK * DINNER * 2);

  // 1) precision conversion / weight transposes
  {
    size_t n = (size_t)NTOK * DMODEL;
    cvt_bf16_kernel<<<(unsigned)((n + 255) / 256), 256, 0, stream>>>(u, ub, n);
  }
  {
    size_t n = (size_t)DMODEL * DPROJ;
    transpose_bf16_kernel<<<(unsigned)((n + 255) / 256), 256, 0, stream>>>(in_w, wtb, DMODEL, DPROJ);
  }
  {
    size_t n = (size_t)DINNER * DMODEL;
    transpose_bf16_kernel<<<(unsigned)((n + 255) / 256), 256, 0, stream>>>(out_w, wtb2, DINNER, DMODEL);
  }

  // 2) in_proj GEMM: zx[8192, 8384] = u[8192,2048] @ W
  gemm_bf16_wmma<<<dim3(DPROJ / 64, NTOK / 64), 128, 0, stream>>>(
      ub, wtb, zx, NTOK, DPROJ, DMODEL);

  // 3) causal conv + silu
  {
    size_t n = (size_t)NTOK * DINNER;
    conv_silu_kernel<<<(unsigned)((n + 255) / 256), 256, 0, stream>>>(zx, conv_w, conv_b, vf, vb);
  }

  // 4) q/k (elu+1) and log-decay
  qkla_kernel<<<NTOK, 64, 0, stream>>>(zx, A_log, dt_bias, qb, kb, la);

  // 5) SSD chunked scan: one block per (batch, head)
  ssd_kernel<<<BATCH * NHEADS, 128, 0, stream>>>(qb, kb, vb, vf, la, D_param, yf);

  // 6) RMSNorm + silu(z) gate -> bf16
  rmsnorm_gate_kernel<<<NTOK, 256, 0, stream>>>(yf, zx, nscale, yb);

  // 7) out_proj GEMM: out[8192,2048] = y[8192,4096] @ Wout
  gemm_bf16_wmma<<<dim3(DMODEL / 64, NTOK / 64), 128, 0, stream>>>(
      yb, wtb2, out, NTOK, DMODEL, DINNER);
}
